// Block_90726889161490
// MI455X (gfx1250) — matrix operand order repaired and checked by reading, not compiled, not run on hardware
//
#include <hip/hip_runtime.h>
#include <math.h>
#include <stdint.h>

typedef __bf16 bf16;
typedef __bf16 bf16x8  __attribute__((ext_vector_type(8)));
typedef __bf16 bf16x16 __attribute__((ext_vector_type(16)));
typedef float  v8f     __attribute__((ext_vector_type(8)));

#define HEADS 16
#define HDIM  64
#define HID   1024
#define BATCH 4
#define SEQ   2048
#define ROWS  (BATCH*SEQ)   /* 8192 */
#define NQKVP (7*HID)       /* 7168 */
#define KCAT  (5*HID)       /* 5120 */

static __device__ __forceinline__ bf16 f2b(float x) { return (bf16)x; }
static __device__ __forceinline__ float b2f(bf16 x) { return (float)x; }

// CDNA5: generic shared-space address low 32 bits == LDS byte address (ISA 10.2 aperture table)
static __device__ __forceinline__ unsigned lds_addr(const void* p) {
  return (unsigned)(uintptr_t)p;
}
// Async global->LDS copy, 16B per lane; tracked by ASYNCcnt (ISA 10.7 / 15.18.3)
static __device__ __forceinline__ void async_to_lds_b128(unsigned lds, const void* gptr) {
  asm volatile("global_load_async_to_lds_b128 %0, %1, off"
               :: "v"(lds), "v"(gptr) : "memory");
}
static __device__ __forceinline__ void wait_async0() {
  asm volatile("s_wait_asynccnt 0" ::: "memory");
}

#define SHUF16(lo, hi) __builtin_shufflevector((lo), (hi), 0,1,2,3,4,5,6,7,8,9,10,11,12,13,14,15)

// ---------------------------------------------------------------- cast f32->bf16
__global__ void cast_f32_bf16(const float* __restrict__ src, bf16* __restrict__ dst, int n) {
  int i = blockIdx.x * blockDim.x + threadIdx.x;
  int stride = gridDim.x * blockDim.x;
  for (; i < n; i += stride) dst[i] = f2b(src[i]);
}

// ---------------------------------------------------------------- LayerNorm -> bf16
__global__ __launch_bounds__(256) void layernorm_bf16(
    const float* __restrict__ X, const float* __restrict__ G,
    const float* __restrict__ Bt, bf16* __restrict__ Xn) {
  __shared__ float2 red[256];
  const int row = blockIdx.x, tid = threadIdx.x;
  const float4 x4 = ((const float4*)(X + (size_t)row * HID))[tid];
  float s = x4.x + x4.y + x4.z + x4.w;
  float q = x4.x*x4.x + x4.y*x4.y + x4.z*x4.z + x4.w*x4.w;
  red[tid] = make_float2(s, q);
  __syncthreads();
  for (int off = 128; off > 0; off >>= 1) {
    if (tid < off) { red[tid].x += red[tid+off].x; red[tid].y += red[tid+off].y; }
    __syncthreads();
  }
  const float mean = red[0].x * (1.0f / HID);
  const float var  = red[0].y * (1.0f / HID) - mean * mean;
  const float rstd = rsqrtf(var + 1e-5f);
  bf16* dst = Xn + (size_t)row * HID + tid * 4;
  const float* gp = G + tid * 4;
  const float* bp = Bt + tid * 4;
  #pragma unroll
  for (int j = 0; j < 4; ++j) {
    float xv = (&x4.x)[j];
    dst[j] = f2b((xv - mean) * rstd * gp[j] + bp[j]);
  }
}

// ---------------------------------------------------------------- bf16 WMMA GEMM
// C[M,N] = A[M,K] * B[K,N]; 128x128 block tile, BK=32, 8 waves of 32x64.
// Tiles staged with async global->LDS copies, double buffered, 1 barrier / K-step.
#define BM 128
#define BN 128
#define BKT 32
#define LDA 48    /* padded LDS row stride (elements), 96B = 16B-aligned */
#define LDB 136   /* padded LDS row stride (elements), 272B = 16B-aligned */

__global__ __launch_bounds__(256) void gemm_bf16_wmma(
    const bf16* __restrict__ A, const bf16* __restrict__ Bm,
    bf16* __restrict__ Cb, float* __restrict__ Cf, const float* __restrict__ bias,
    int M, int N, int K) {
  __shared__ bf16 As[2][BM * LDA];
  __shared__ bf16 Bs[2][BKT * LDB];
  const int tid  = threadIdx.x;
  const int lane = tid & 31;
  const int wave = tid >> 5;       // 0..7
  const int wm   = wave & 3;       // wave position in M (4)
  const int wn   = wave >> 2;      // wave position in N (2)
  const int hl   = lane >> 4;      // half-wave
  const int l16  = lane & 15;
  const int m0   = blockIdx.y * BM;
  const int n0   = blockIdx.x * BN;

  // per-thread async copy slots: A tile 128x32 = 512 vec8, B tile 32x128 = 512 vec8
  const int rA0 = tid >> 2,         cA0 = (tid & 3) * 8;
  const int rA1 = (tid + 256) >> 2, cA1 = ((tid + 256) & 3) * 8;
  const int rB0 = tid >> 4,         cB0 = (tid & 15) * 8;
  const int rB1 = (tid + 256) >> 4, cB1 = ((tid + 256) & 15) * 8;
  const bf16* gA0 = A + (size_t)(m0 + rA0) * K + cA0;
  const bf16* gA1 = A + (size_t)(m0 + rA1) * K + cA1;
  const bf16* gB0 = Bm + (size_t)rB0 * N + n0 + cB0;
  const bf16* gB1 = Bm + (size_t)rB1 * N + n0 + cB1;
  unsigned lA0[2], lA1[2], lB0[2], lB1[2];
  #pragma unroll
  for (int b = 0; b < 2; ++b) {
    lA0[b] = lds_addr(&As[b][rA0 * LDA + cA0]);
    lA1[b] = lds_addr(&As[b][rA1 * LDA + cA1]);
    lB0[b] = lds_addr(&Bs[b][rB0 * LDB + cB0]);
    lB1[b] = lds_addr(&Bs[b][rB1 * LDB + cB1]);
  }
  auto issue = [&](int buf, int kt) {
    const size_t ka = (size_t)kt * BKT;        // A advances BKT cols
    const size_t kb = (size_t)kt * BKT * N;    // B advances BKT rows
    async_to_lds_b128(lA0[buf], gA0 + ka);
    async_to_lds_b128(lA1[buf], gA1 + ka);
    async_to_lds_b128(lB0[buf], gB0 + kb);
    async_to_lds_b128(lB1[buf], gB1 + kb);
  };

  v8f acc[2][4] = {};
  const int nk = K / BKT;
  issue(0, 0);
  for (int kt = 0; kt < nk; ++kt) {
    const int cur = kt & 1;
    wait_async0();        // own 4 copies for buffer `cur` have landed in LDS
    __syncthreads();      // everyone's copies landed; everyone done reading buffer cur^1
    if (kt + 1 < nk) issue(cur ^ 1, kt + 1);  // async engine runs under compute

    #pragma unroll
    for (int mt = 0; mt < 2; ++mt) {
      const bf16* ap = &As[cur][(wm * 32 + mt * 16 + l16) * LDA];
      bf16x8 alo = *(const bf16x8*)(ap + hl * 8);
      bf16x8 ahi = *(const bf16x8*)(ap + 16 + hl * 8);
      bf16x16 afrag = SHUF16(alo, ahi);
      #pragma unroll
      for (int nt = 0; nt < 4; ++nt) {
        const bf16* bp = &Bs[cur][wn * 64 + nt * 16 + l16];
        bf16x16 bfrag;
        #pragma unroll
        for (int s = 0; s < 16; ++s) bfrag[s] = bp[((s & 7) + 8 * hl + 16 * (s >> 3)) * LDB];
        acc[mt][nt] = __builtin_amdgcn_wmma_f32_16x16x32_bf16(
            false, afrag, false, bfrag, (short)0, acc[mt][nt], false, false);
      }
    }
  }

  #pragma unroll
  for (int mt = 0; mt < 2; ++mt)
    #pragma unroll
    for (int nt = 0; nt < 4; ++nt)
      #pragma unroll
      for (int r = 0; r < 8; ++r) {
        int m = m0 + wm * 32 + mt * 16 + r + 8 * hl;
        int n = n0 + wn * 64 + nt * 16 + l16;
        float vv = acc[mt][nt][r];
        if (Cb) Cb[(size_t)m * N + n] = f2b(vv);
        else    Cf[(size_t)m * N + n] = vv + bias[n];
      }
}

// ---------------------------------------------------------------- rotary / gelu / layout
// qkvp[row,7168] -> Q[B,H,T,64]*0.125 rot, Kt[B,H,64,T] rot, V[B,H,T,64], gelu(p)->cat cols 1024..5119
__global__ __launch_bounds__(512) void postproc(
    const bf16* __restrict__ qkvp, bf16* __restrict__ Q, bf16* __restrict__ Kt,
    bf16* __restrict__ V, bf16* __restrict__ Cat) {
  const int row = blockIdx.x;
  const int b = row >> 11;        // /2048
  const int t = row & 2047;
  const int tid = threadIdx.x;    // 0..511
  const bf16* src = qkvp + (size_t)row * NQKVP;
  const int h = tid >> 5, i = tid & 31;
  const float inv = __expf(-(float)i * (9.210340371976184f / 32.0f)); // 10000^(-i/32)
  float sn, cs;
  __sincosf((float)t * inv, &sn, &cs);
  {
    float x1 = b2f(src[h * 64 + i]);
    float x2 = b2f(src[h * 64 + 32 + i]);
    bf16* qd = Q + ((size_t)(b * HEADS + h) * SEQ + t) * HDIM;
    qd[i]      = f2b((x1 * cs - x2 * sn) * 0.125f);   // fold HEAD_DIM^-0.5
    qd[32 + i] = f2b((x1 * sn + x2 * cs) * 0.125f);
  }
  {
    float x1 = b2f(src[HID + h * 64 + i]);
    float x2 = b2f(src[HID + h * 64 + 32 + i]);
    bf16* kd = Kt + (size_t)(b * HEADS + h) * HDIM * SEQ;
    kd[(size_t)i * SEQ + t]        = f2b(x1 * cs - x2 * sn);
    kd[(size_t)(32 + i) * SEQ + t] = f2b(x1 * sn + x2 * cs);
  }
  #pragma unroll
  for (int j = tid; j < HID; j += 512) {
    int h2 = j >> 6, d = j & 63;
    V[((size_t)(b * HEADS + h2) * SEQ + t) * HDIM + d] = src[2 * HID + j];
  }
  bf16* oc = Cat + (size_t)row * KCAT + HID;
  for (int j = tid; j < 4 * HID; j += 512) {
    float x = b2f(src[3 * HID + j]);
    oc[j] = f2b(0.5f * x * (1.0f + erff(x * 0.70710678f)));  // exact gelu
  }
}

// ---------------------------------------------------------------- flash attention (WMMA)
// block = 8 waves; each wave owns 16 query rows, streams keys in chunks of 64.
#define PLD 72   /* per-wave P tile LDS row stride: 144B, 16B-aligned */
__global__ __launch_bounds__(256) void attn_wmma(
    const bf16* __restrict__ Q,    // [B,H,T,64]
    const bf16* __restrict__ Kt,   // [B,H,64,T]
    const bf16* __restrict__ V,    // [B,H,T,64]
    bf16* __restrict__ Cat) {      // [ROWS,5120], o -> cols 0..1023
  __shared__ bf16 Pl[8 * 16 * PLD];
  const int tid = threadIdx.x, lane = tid & 31, wave = tid >> 5;
  const int hl = lane >> 4, l16 = lane & 15;
  const int qb = blockIdx.x & 15;
  const int hh = (blockIdx.x >> 4) & 15;
  const int bb = blockIdx.x >> 8;
  const int q0 = qb * 128 + wave * 16;
  const bf16* qp = Q + ((size_t)(bb * HEADS + hh) * SEQ + q0) * HDIM;
  const bf16* kp = Kt + (size_t)(bb * HEADS + hh) * HDIM * SEQ;
  const bf16* vp = V + (size_t)(bb * HEADS + hh) * SEQ * HDIM;
  bf16* pl = &Pl[wave * 16 * PLD];

  bf16x16 qf[2];
  #pragma unroll
  for (int c = 0; c < 2; ++c) {
    const bf16* p = qp + l16 * HDIM + c * 32;
    bf16x8 lo = *(const bf16x8*)(p + hl * 8);
    bf16x8 hi = *(const bf16x8*)(p + 16 + hl * 8);
    qf[c] = SHUF16(lo, hi);
  }

  float mrow[8], lrow[8], alpha[8];
  v8f o[4] = {};
  #pragma unroll
  for (int r = 0; r < 8; ++r) { mrow[r] = -1e30f; lrow[r] = 0.0f; }

  for (int j0 = 0; j0 < SEQ; j0 += 64) {
    // S = Q * K^T for this 16x64 tile
    v8f s[4] = {};
    #pragma unroll
    for (int nt = 0; nt < 4; ++nt)
      #pragma unroll
      for (int c = 0; c < 2; ++c) {
        const bf16* p = kp + (size_t)(c * 32) * SEQ + j0 + nt * 16 + l16;
        bf16x16 bfrag;
        #pragma unroll
        for (int s = 0; s < 16; ++s) bfrag[s] = p[(size_t)((s & 7) + 8 * hl + 16 * (s >> 3)) * SEQ];
        s[nt] = __builtin_amdgcn_wmma_f32_16x16x32_bf16(
            false, qf[c], false, bfrag, (short)0, s[nt], false, false);
      }
    // online softmax: rows live in (r, half); reduce across 16 lanes
    #pragma unroll
    for (int r = 0; r < 8; ++r) {
      float tmx = fmaxf(fmaxf(s[0][r], s[1][r]), fmaxf(s[2][r], s[3][r]));
      #pragma unroll
      for (int msk = 1; msk < 16; msk <<= 1) tmx = fmaxf(tmx, __shfl_xor(tmx, msk, 32));
      float mn = fmaxf(mrow[r], tmx);
      alpha[r] = __expf(mrow[r] - mn);
      mrow[r] = mn;
    }
    #pragma unroll
    for (int r = 0; r < 8; ++r) {
      float rs = 0.0f;
      #pragma unroll
      for (int nt = 0; nt < 4; ++nt) {
        float pv = __expf(s[nt][r] - mrow[r]);
        s[nt][r] = pv;
        rs += pv;
      }
      #pragma unroll
      for (int msk = 1; msk < 16; msk <<= 1) rs += __shfl_xor(rs, msk, 32);
      lrow[r] = lrow[r] * alpha[r] + rs;
      #pragma unroll
      for (int dt = 0; dt < 4; ++dt) o[dt][r] *= alpha[r];
    }
    // C-layout P -> LDS -> A-layout fragments (per-wave region; DS in-order per wave)
    #pragma unroll
    for (int nt = 0; nt < 4; ++nt)
      #pragma unroll
      for (int r = 0; r < 8; ++r)
        pl[(r + 8 * hl) * PLD + nt * 16 + l16] = f2b(s[nt][r]);
    bf16x16 pf[2];
    #pragma unroll
    for (int c = 0; c < 2; ++c) {
      const bf16* p = pl + l16 * PLD + c * 32;
      bf16x8 lo = *(const bf16x8*)(p + hl * 8);
      bf16x8 hi = *(const bf16x8*)(p + 16 + hl * 8);
      pf[c] = SHUF16(lo, hi);
    }
    // O += P * V
    #pragma unroll
    for (int dt = 0; dt < 4; ++dt)
      #pragma unroll
      for (int c = 0; c < 2; ++c) {
        const bf16* p = vp + (size_t)(j0 + c * 32) * HDIM + dt * 16 + l16;
        bf16x16 vfrag;
        #pragma unroll
        for (int s = 0; s < 16; ++s) vfrag[s] = p[(size_t)((s & 7) + 8 * hl + 16 * (s >> 3)) * HDIM];
        o[dt] = __builtin_amdgcn_wmma_f32_16x16x32_bf16(
            false, pf[c], false, vfrag, (short)0, o[dt], false, false);
      }
  }
  #pragma unroll
  for (int dt = 0; dt < 4; ++dt)
    #pragma unroll
    for (int r = 0; r < 8; ++r) {
      int t = q0 + r + 8 * hl;
      int col = hh * HDIM + dt * 16 + l16;
      Cat[(size_t)(bb * SEQ + t) * KCAT + col] = f2b(o[dt][r] / lrow[r]);
    }
}

// ---------------------------------------------------------------- launcher
extern "C" void kernel_launch(void* const* d_in, const int* in_sizes, int n_in,
                              void* d_out, int out_size, void* d_ws, size_t ws_size,
                              hipStream_t stream) {
  (void)in_sizes; (void)n_in; (void)out_size; (void)ws_size;
  const float* x     = (const float*)d_in[0];
  const float* gam   = (const float*)d_in[1];
  const float* bet   = (const float*)d_in[2];
  const float* w_in  = (const float*)d_in[3];
  const float* w_out = (const float*)d_in[4];
  const float* b_out = (const float*)d_in[5];
  float* out = (float*)d_out;

  char* ws = (char*)d_ws;
  size_t off = 0;
  auto alloc = [&](size_t bytes) -> void* {
    void* p = ws + off;
    off += (bytes + 255) & ~(size_t)255;
    return p;
  };
  bf16* xn    = (bf16*)alloc((size_t)ROWS * HID * 2);    // 16 MB
  bf16* winb  = (bf16*)alloc((size_t)HID * NQKVP * 2);   // 14 MB
  bf16* woutb = (bf16*)alloc((size_t)KCAT * HID * 2);    // 10 MB
  bf16* qkvp  = (bf16*)alloc((size_t)ROWS * NQKVP * 2);  // 112 MB
  bf16* Q     = (bf16*)alloc((size_t)ROWS * HID * 2);    // 16 MB
  bf16* Kt    = (bf16*)alloc((size_t)ROWS * HID * 2);    // 16 MB
  bf16* V     = (bf16*)alloc((size_t)ROWS * HID * 2);    // 16 MB
  bf16* cat   = (bf16*)alloc((size_t)ROWS * KCAT * 2);   // 80 MB

  cast_f32_bf16<<<2048, 256, 0, stream>>>(w_in, winb, HID * NQKVP);
  cast_f32_bf16<<<2048, 256, 0, stream>>>(w_out, woutb, KCAT * HID);
  layernorm_bf16<<<ROWS, 256, 0, stream>>>(x, gam, bet, xn);
  gemm_bf16_wmma<<<dim3(NQKVP / BN, ROWS / BM), 256, 0, stream>>>(
      xn, winb, qkvp, nullptr, nullptr, ROWS, NQKVP, HID);
  postproc<<<ROWS, 512, 0, stream>>>(qkvp, Q, Kt, V, cat);
  attn_wmma<<<BATCH * HEADS * (SEQ / 128), 256, 0, stream>>>(Q, Kt, V, cat);
  gemm_bf16_wmma<<<dim3(HID / BN, ROWS / BM), 256, 0, stream>>>(
      cat, woutb, nullptr, out, b_out, ROWS, HID, KCAT);
}
